// RGLRU_36481452213170
// MI455X (gfx1250) — matrix operand order repaired and checked by reading, not compiled, not run on hardware
//
#include <hip/hip_runtime.h>
#include <math.h>

typedef __bf16 bf16;
typedef __attribute__((ext_vector_type(16))) __bf16 v16bf;
typedef __attribute__((ext_vector_type(8)))  float  v8f;
typedef __attribute__((ext_vector_type(8)))  float  float8;

#define B_   8
#define T_   4096
#define W_   1024
#define H_   8
#define BW_  128
#define CL_  64        // tokens per block == scan chunk length
#define CH_  64        // number of chunks (CH_*CL_ == T_)
#define LDP  (BW_ + 4) // padded LDS row stride (floats)

// ---------------------------------------------------------------------------
// Kernel 0: weight transpose+bf16 convert, and 8*softplus(a_param)
//   w[h][i][j] (f32) -> wt[h][j][i] (bf16)  (B-fragments become contiguous)
// ---------------------------------------------------------------------------
__global__ __launch_bounds__(256) void rglru_prep(
    const float* __restrict__ w_in, const float* __restrict__ w_a,
    const float* __restrict__ a_param,
    bf16* __restrict__ wt_in, bf16* __restrict__ wt_a,
    float* __restrict__ c8)
{
  int idx = blockIdx.x * 256 + threadIdx.x;
  if (idx < H_ * BW_ * BW_) {
    int h = idx / (BW_ * BW_);
    int r = idx % (BW_ * BW_);
    int i = r / BW_;
    int j = r % BW_;
    wt_in[(h * BW_ + j) * BW_ + i] = (bf16)w_in[idx];
    wt_a [(h * BW_ + j) * BW_ + i] = (bf16)w_a [idx];
  }
  if (idx < W_) {
    float v  = a_param[idx];
    float sp = fmaxf(v, 0.f) + log1pf(__expf(-fabsf(v)));  // stable softplus
    c8[idx]  = 8.f * sp;
  }
}

// ---------------------------------------------------------------------------
// Fused gates (bf16 WMMA) + chunk-local scan. One block = one (b, h, chunk)
// of 64 tokens x 128 channels. 2 waves; each wave owns TWO 16-token M-tiles,
// so every B-fragment load from L2 feeds 4 WMMA chains (2 M-tiles x 2 gate
// matrices). Gates land in LDS only; the block then scans the chunk.
//   FINAL=false: emit per-chunk decay product Pc and final state Hc (NT).
//   FINAL=true : recompute gates, apply chunk carry Cin, write y (NT).
// ---------------------------------------------------------------------------
template <bool FINAL>
__global__ __launch_bounds__(64) void rglru_gates_scan(
    const float* __restrict__ x,
    const float* __restrict__ b_in, const float* __restrict__ b_a,
    const bf16* __restrict__ wt_in, const bf16* __restrict__ wt_a,
    const float* __restrict__ c8,
    const float* __restrict__ Cin,
    float* __restrict__ Pc, float* __restrict__ Hc,
    float* __restrict__ y)
{
  __shared__ float a_lds [CL_][LDP];
  __shared__ float nx_lds[CL_][LDP];

  const int lane = threadIdx.x & 31;
  const int wv   = threadIdx.x >> 5;   // 0..1
  const int tch  = blockIdx.x;         // chunk 0..63
  const int h    = blockIdx.y;
  const int b    = blockIdx.z;

  const int mrow = lane & 15;          // token row for A fragment
  const int kh   = lane >> 4;          // K-half select
  const int t0   = tch * CL_ + wv * 32;  // this wave's 32 tokens (2 M-tiles)

  // ---- bf16 A fragments (16-bit 16x32 layout): lane half kh holds
  // K in [kh*16,+16) — two contiguous 32B chunks of the x row
  // (matching the B fragment K order), converted in registers.
  v16bf afrag[2][4];
#pragma unroll
  for (int mt = 0; mt < 2; ++mt) {
    const size_t rowbase =
        ((size_t)(b * T_) + t0 + mt * 16 + mrow) * W_ + h * BW_;
#pragma unroll
    for (int ks = 0; ks < 4; ++ks) {
      const float8 c1 = *(const float8*)(x + rowbase + ks * 32 + kh * 16);
      const float8 c2 = *(const float8*)(x + rowbase + ks * 32 + kh * 16 + 8);
      v16bf f;
#pragma unroll
      for (int i = 0; i < 8; ++i) { f[i] = (bf16)c1[i]; f[i + 8] = (bf16)c2[i]; }
      afrag[mt][ks] = f;
    }
  }

#pragma unroll 1
  for (int nt = 0; nt < 8; ++nt) {
    const int n = nt * 16 + (lane & 15);
    const bf16* wrx = wt_in + (size_t)(h * BW_ + n) * BW_;
    const bf16* wra = wt_a  + (size_t)(h * BW_ + n) * BW_;
    if (nt < 7) {                       // gfx1250 global_prefetch_b8
      __builtin_prefetch(wrx + BW_, 0, 0);
      __builtin_prefetch(wra + BW_, 0, 0);
    }

    // B fragments (32x16 bf16): 16 contiguous K values for column n.
    v16bf bx[4], ba[4];
#pragma unroll
    for (int ks = 0; ks < 4; ++ks) {
      bx[ks] = *(const v16bf*)(wrx + ks * 32 + kh * 16);
      ba[ks] = *(const v16bf*)(wra + ks * 32 + kh * 16);
    }

    v8f accx[2] = {{}, {}};
    v8f acca[2] = {{}, {}};
#pragma unroll
    for (int ks = 0; ks < 4; ++ks) {
#pragma unroll
      for (int mt = 0; mt < 2; ++mt) {   // reuse bx/ba across both M-tiles
        accx[mt] = __builtin_amdgcn_wmma_f32_16x16x32_bf16(
            false, afrag[mt][ks], false, bx[ks], (short)0, accx[mt], false, false);
        acca[mt] = __builtin_amdgcn_wmma_f32_16x16x32_bf16(
            false, afrag[mt][ks], false, ba[ks], (short)0, acca[mt], false, false);
      }
    }

    const float bi  = b_in[h * BW_ + n];
    const float bav = b_a [h * BW_ + n];
    const float c8v = c8  [h * BW_ + n];

    // D layout: VGPR r -> (M = r + 8*(lane>=16), N = lane%16)
#pragma unroll
    for (int mt = 0; mt < 2; ++mt) {
#pragma unroll
      for (int r = 0; r < 8; ++r) {
        const int m    = r + ((lane >= 16) ? 8 : 0);
        const int tloc = wv * 32 + mt * 16 + m;   // token within chunk
        const int tok  = tch * CL_ + tloc;        // global token
        const size_t oidx = ((size_t)(b * T_) + tok) * W_ + h * BW_ + n;

        const float zx = accx[mt][r] + bi;
        const float za = acca[mt][r] + bav;
        const float gx = 1.f / (1.f + __expf(-zx));
        const float ga = 1.f / (1.f + __expf(-za));
        const float la = -ga * c8v;                  // log_a
        const float av = __expf(la);
        // exp(2*log_a) == av*av  -> one less transcendental
        const float mu = sqrtf(fmaxf(0.f, 1.f - av * av));
        const float xv = x[oidx];                    // L2-hot reload
        const bool  fr = (tok == 0);

        a_lds [tloc][n] = fr ? 0.f : av;
        nx_lds[tloc][n] = xv * gx * (fr ? 1.f : mu);
      }
    }
  }

  __syncthreads();

  // ---- chunk-local scan: thread owns channels ch and ch+64.
  const int ch = threadIdx.x;          // 0..63
  const size_t cidx = ((size_t)(b * CH_) + tch) * W_ + h * BW_ + ch;

  float carry0 = 0.f, carry1 = 0.f;
  if (FINAL) { carry0 = Cin[cidx]; carry1 = Cin[cidx + 64]; }

  float h0 = 0.f, p0 = 1.f, h1 = 0.f, p1 = 1.f;
  size_t base = ((size_t)b * T_ + (size_t)tch * CL_) * W_ + h * BW_ + ch;
#pragma unroll 4
  for (int t = 0; t < CL_; ++t) {
    const float av0 = a_lds [t][ch];
    const float nx0 = nx_lds[t][ch];
    const float av1 = a_lds [t][ch + 64];
    const float nx1 = nx_lds[t][ch + 64];
    h0 = fmaf(av0, h0, nx0);
    h1 = fmaf(av1, h1, nx1);
    p0 *= av0;
    p1 *= av1;
    if (FINAL) {
      __builtin_nontemporal_store(fmaf(p0, carry0, h0), y + base);
      __builtin_nontemporal_store(fmaf(p1, carry1, h1), y + base + 64);
    }
    base += W_;
  }
  if (!FINAL) {
    __builtin_nontemporal_store(p0, Pc + cidx);
    __builtin_nontemporal_store(p1, Pc + cidx + 64);
    __builtin_nontemporal_store(h0, Hc + cidx);
    __builtin_nontemporal_store(h1, Hc + cidx + 64);
  }
}

// ---------------------------------------------------------------------------
// Carry combine across the 64 chunks (tiny, sequential over chunks).
// Cin[c] = h entering chunk c.
// ---------------------------------------------------------------------------
__global__ __launch_bounds__(256) void rglru_scan_carry(
    const float* __restrict__ Pc, const float* __restrict__ Hc,
    float* __restrict__ Cin)
{
  const int idx = blockIdx.x * 256 + threadIdx.x;  // b*W_ + w
  const int w = idx % W_;
  const int b = idx / W_;
  float carry = 0.f;
#pragma unroll 4
  for (int c = 0; c < CH_; ++c) {
    const size_t i = ((size_t)b * CH_ + c) * W_ + w;
    __builtin_nontemporal_store(carry, Cin + i);
    carry = fmaf(__builtin_nontemporal_load(Pc + i), carry,
                 __builtin_nontemporal_load(Hc + i));
  }
}

// ---------------------------------------------------------------------------
extern "C" void kernel_launch(void* const* d_in, const int* in_sizes, int n_in,
                              void* d_out, int out_size, void* d_ws, size_t ws_size,
                              hipStream_t stream)
{
  const float* x       = (const float*)d_in[0];
  const float* w_in    = (const float*)d_in[1];
  const float* b_in    = (const float*)d_in[2];
  const float* w_a     = (const float*)d_in[3];
  const float* b_a     = (const float*)d_in[4];
  const float* a_param = (const float*)d_in[5];
  float* y = (float*)d_out;

  char* ws = (char*)d_ws;
  // Workspace (bytes): wt_in [0,256K), wt_a [256K,512K), c8 [512K,+4K),
  // then Pc, Hc, Cin : f32 [B][CH][W] (2 MiB apiece). Total 7 MiB.
  bf16*  wt_in = (bf16*)(ws);
  bf16*  wt_a  = (bf16*)(ws + (256u << 10));
  float* c8    = (float*)(ws + (512u << 10));
  float* Pc    = (float*)(ws + (1u << 20));
  float* Hc    = Pc + (size_t)B_ * CH_ * W_;
  float* Cin   = Hc + (size_t)B_ * CH_ * W_;

  rglru_prep<<<(H_ * BW_ * BW_) / 256, 256, 0, stream>>>(
      w_in, w_a, a_param, wt_in, wt_a, c8);

  const dim3 grid(T_ / CL_, H_, B_);
  rglru_gates_scan<false><<<grid, 64, 0, stream>>>(
      x, b_in, b_a, wt_in, wt_a, c8, nullptr, Pc, Hc, nullptr);

  rglru_scan_carry<<<(B_ * W_) / 256, 256, 0, stream>>>(Pc, Hc, Cin);

  rglru_gates_scan<true><<<grid, 64, 0, stream>>>(
      x, b_in, b_a, wt_in, wt_a, c8, Cin, nullptr, nullptr, y);
}
